// SudokuSolver_72825465470990
// MI455X (gfx1250) — matrix operand order repaired and checked by reading, not compiled, not run on hardware
//
#include <hip/hip_runtime.h>

typedef __attribute__((ext_vector_type(16))) _Float16 v16h;
typedef __attribute__((ext_vector_type(8)))  _Float16 v8h;
typedef __attribute__((ext_vector_type(8)))  float    v8f;

union V16 { v16h v; v8h h8[2]; };

#define THREADS 128

// One workgroup per board. 4 waves (wave32). All state LDS-resident.
__global__ __launch_bounds__(THREADS) void sudoku_wmma_kernel(
    const float* __restrict__ x,   // [B,81,9]
    const float* __restrict__ W1,  // [128,27]
    const float* __restrict__ W2,  // [9,128]
    float* __restrict__ out,       // [B,81,9]
    int B)
{
    __shared__ float     s_xc[81 * 9];        // current board (one-hot / empty)
    __shared__ float     s_xp[81 * 9];        // soft predictions (output)
    __shared__ float     s_sums[244];         // [type][idx][digit] + zero slot [243]
    __shared__ _Float16  s_fA[6 * 32 * 16];   // GEMM1 A matrix in WMMA fragment order
    __shared__ _Float16  s_W1B[32 * 128];     // W1^T, K-major, padded K 27->32
    __shared__ _Float16  s_W2T[128 * 16];     // W2^T, padded N 9->16
    __shared__ _Float16  s_h[96 * 128];       // hidden, padded M 81->96
    __shared__ float     s_y[96 * 16];        // logits
    __shared__ float     s_score[81];
    __shared__ int       s_pos[81];

    const int tid  = threadIdx.x;
    const int lane = tid & 31;
    const int wv   = tid >> 5;            // 0..3
    const int b    = blockIdx.x;
    if (b >= B) return;

    __builtin_prefetch(W1, 0, 3);
    __builtin_prefetch(W2, 0, 3);
    __builtin_prefetch(x + b * 729, 0, 3);

    // ---- init: load board + weights into LDS ----
    for (int i = tid; i < 81 * 9; i += THREADS) {
        float v = x[b * 729 + i];
        s_xc[i] = v;
        s_xp[i] = v;
    }
    for (int i = tid; i < 32 * 128; i += THREADS) {
        int k = i >> 7, n = i & 127;
        s_W1B[i] = (k < 27) ? (_Float16)W1[n * 27 + k] : (_Float16)0.0f;
    }
    for (int i = tid; i < 128 * 16; i += THREADS) {
        int k = i >> 4, n = i & 15;
        s_W2T[i] = (n < 9) ? (_Float16)W2[n * 128 + k] : (_Float16)0.0f;
    }
    if (tid == 0) s_sums[243] = 0.0f;     // permanent zero slot for padded gathers
    __syncthreads();

    // ---- build loop-invariant B fragments in registers ----
    // GEMM1 B (27x128 -> 32x128), this wave owns N-tiles 2*wv and 2*wv+1.
    v16h b1lo, b1hi;
    {
        int col0 = (2 * wv) * 16 + (lane & 15);
        int kb   = (lane < 16) ? 0 : 8;
#pragma unroll
        for (int i = 0; i < 16; ++i) {
            b1lo[i] = s_W1B[(kb + ((i < 8) ? i : (8 + i))) * 128 + col0];
            b1hi[i] = s_W1B[(kb + ((i < 8) ? i : (8 + i))) * 128 + col0 + 16];
        }
    }
    // GEMM2 B (128x16), 4 K-steps.
    v16h b2[4];
    {
        int col = lane & 15;
#pragma unroll
        for (int ks = 0; ks < 4; ++ks) {
            int kb = ks * 32 + ((lane < 16) ? 0 : 8);
#pragma unroll
            for (int i = 0; i < 16; ++i)
                b2[ks][i] = s_W2T[(kb + ((i < 8) ? i : (8 + i))) * 16 + col];
        }
    }

    // ---- 81 sequential propagation steps ----
    for (int it = 0; it < 81; ++it) {
        // (1) constraint digit counts: 243 sums of 9
        for (int o = tid; o < 243; o += THREADS) {
            int t   = o / 81;
            int rem = o - t * 81;
            int idx = rem / 9;
            int d   = rem - idx * 9;
            float s = 0.0f;
            if (t == 0) {
#pragma unroll
                for (int j = 0; j < 9; ++j) s += s_xc[(idx * 9 + j) * 9 + d];
            } else if (t == 1) {
#pragma unroll
                for (int j = 0; j < 9; ++j) s += s_xc[(idx + 9 * j) * 9 + d];
            } else {
                int base = 27 * (idx / 3) + 3 * (idx % 3);
#pragma unroll
                for (int j = 0; j < 9; ++j)
                    s += s_xc[(base + 9 * (j / 3) + (j % 3)) * 9 + d];
            }
            s_sums[o] = s;
        }
        __syncthreads();

        // (1.5) stage GEMM1 A matrix in fragment order (branch-free gather).
        // Fragment row fr = mt*32 + laneSlot: 16 contiguous f16.
        for (int fr = tid; fr < 192; fr += THREADS) {
            int mt  = fr >> 5;
            int l   = fr & 31;
            int row = mt * 16 + (l & 15);
            int kb  = (l < 16) ? 0 : 8;
            int r   = row / 9;
            int cI  = row % 9;
            int bx  = 3 * (row / 27) + cI / 3;
            bool rowok = row < 81;
            v16h vals;
#pragma unroll
            for (int i = 0; i < 16; ++i) {
                int K = kb + ((i < 8) ? i : (8 + i));   // K 0-7,16-23 | 8-15,24-31
                int t = K / 9, d = K - 9 * t;
                int idx  = (t == 0) ? r : ((t == 1) ? cI : bx);
                int addr = (rowok && K < 27) ? (t * 81 + idx * 9 + d) : 243;
                vals[i] = (_Float16)s_sums[addr];       // unconditional LDS load
            }
            *(v16h*)(s_fA + fr * 16) = vals;            // 2x ds_store_b128
        }
        __syncthreads();

        // (2) GEMM1: h = relu(f @ W1^T)   [96x32]x[32x128]
        for (int mt = 0; mt < 6; ++mt) {
            v16h a = *(const v16h*)(s_fA + (mt * 32 + lane) * 16);  // 2x ds_load_b128
            v8f c0 = {};
            c0 = __builtin_amdgcn_wmma_f32_16x16x32_f16(false, a, false, b1lo,
                                                        (short)0, c0, false, false);
            v8f c1 = {};
            c1 = __builtin_amdgcn_wmma_f32_16x16x32_f16(false, a, false, b1hi,
                                                        (short)0, c1, false, false);
            int n0 = (2 * wv) * 16 + (lane & 15);
            int mb = mt * 16 + ((lane < 16) ? 0 : 8);
#pragma unroll
            for (int v2 = 0; v2 < 8; ++v2) {
                float h0 = c0[v2] > 0.0f ? c0[v2] : 0.0f;
                float h1 = c1[v2] > 0.0f ? c1[v2] : 0.0f;
                s_h[(mb + v2) * 128 + n0]      = (_Float16)h0;
                s_h[(mb + v2) * 128 + n0 + 16] = (_Float16)h1;
            }
        }
        __syncthreads();

        // (3) GEMM2: y = h @ W2^T   [96x128]x[128x16]
        for (int mt = wv; mt < 6; mt += 4) {
            v8f c = {};
#pragma unroll
            for (int ks = 0; ks < 4; ++ks) {
                int row = mt * 16 + (lane & 15);
                int kb  = ks * 32 + ((lane < 16) ? 0 : 8);
                V16 a;
                a.h8[0] = *(const v8h*)(s_h + row * 128 + kb);        // ds_load_b128
                a.h8[1] = *(const v8h*)(s_h + row * 128 + kb + 16);   // ds_load_b128
                c = __builtin_amdgcn_wmma_f32_16x16x32_f16(false, a.v, false, b2[ks],
                                                           (short)0, c, false, false);
            }
            int n  = lane & 15;
            int mb = mt * 16 + ((lane < 16) ? 0 : 8);
#pragma unroll
            for (int v2 = 0; v2 < 8; ++v2)
                s_y[(mb + v2) * 16 + n] = c[v2];
        }
        __syncthreads();

        // (4) per-cell softmax, xp update, per-cell score/argmax
        if (tid < 81) {
            int cell = tid;
            float occ = 0.0f;
#pragma unroll
            for (int d = 0; d < 9; ++d) occ += s_xc[cell * 9 + d];
            bool empty = (occ == 0.0f);

            float yv[9];
#pragma unroll
            for (int d = 0; d < 9; ++d) yv[d] = s_y[cell * 16 + d];
            float m = yv[0];
#pragma unroll
            for (int d = 1; d < 9; ++d) m = fmaxf(m, yv[d]);
            float e[9], ssum = 0.0f;
#pragma unroll
            for (int d = 0; d < 9; ++d) { e[d] = __expf(yv[d] - m); ssum += e[d]; }
            float inv = 1.0f / ssum;

            float best = -1.0f; int bp = 0;
#pragma unroll
            for (int d = 0; d < 9; ++d) {
                float p = e[d] * inv;
                if (empty) s_xp[cell * 9 + d] = p;
                if (p > best) { best = p; bp = d; }   // first-max tie-break
            }
            s_score[cell] = empty ? best : 0.0f;
            s_pos[cell]   = empty ? bp : 0;
        }
        __syncthreads();

        // (5) board-wide most-confident empty cell + hard scatter
        if (tid == 0) {
            float best = s_score[0]; int bc = 0;
            for (int cl = 1; cl < 81; ++cl) {
                float sc = s_score[cl];
                if (sc > best) { best = sc; bc = cl; }  // first-max tie-break
            }
            if (best > 0.0f)                            // == has_empty
                s_xc[bc * 9 + s_pos[bc]] = 1.0f;
        }
        __syncthreads();
    }

    // ---- write xp out ----
    for (int i = tid; i < 81 * 9; i += THREADS)
        out[b * 729 + i] = s_xp[i];
}

extern "C" void kernel_launch(void* const* d_in, const int* in_sizes, int n_in,
                              void* d_out, int out_size, void* d_ws, size_t ws_size,
                              hipStream_t stream) {
    const float* x  = (const float*)d_in[0];
    const float* W1 = (const float*)d_in[1];
    const float* W2 = (const float*)d_in[2];
    float* out = (float*)d_out;
    int B = in_sizes[0] / 729;
    sudoku_wmma_kernel<<<dim3(B), dim3(THREADS), 0, stream>>>(x, W1, W2, out, B);
}
